// ResASPP2_train_57715770524154
// MI455X (gfx1250) — matrix operand order repaired and checked by reading, not compiled, not run on hardware
//
#include <hip/hip_runtime.h>
#include <hip/hip_bf16.h>

typedef __attribute__((ext_vector_type(16))) __bf16 bf16x16;
typedef __attribute__((ext_vector_type(8)))  float  f32x8;

#define B_  8
#define H_  256
#define W_  256
#define HW  65536
#define EPS 1e-5f

// ---------------------------------------------------------------------------
// One-shot weight packer: OIHW f32 -> bf16 A-fragments in WMMA lane order, k contiguous per lane half (same as the B tiles).
// Fragment (mt, kc): 32 lanes x 16 bf16. Lane l (hi=l>>4, lo=l&15) holds
// row m = mt*16+lo, elements j: kk = kc*32 + (hi<<4) + (j<<1).
// Cout-mask and K-tail zero-padding resolved here, once.
// ---------------------------------------------------------------------------
__global__ __launch_bounds__(256) void pack_w(
    const float* __restrict__ wgt, unsigned int* __restrict__ apack,
    int Cout, int K, int nkc, int total)
{
    int tid = blockIdx.x * blockDim.x + threadIdx.x;
    if (tid >= total) return;
    int lane = tid & 31;
    int kc   = (tid >> 5) % nkc;
    int mt   = tid / (32 * nkc);
    int hi   = (lane >> 4) & 1;
    int lo   = lane & 15;
    int m    = mt * 16 + lo;
    bool mv  = (m < Cout);
    const float* wrow = wgt + (size_t)m * K;
    unsigned int o[8];
#pragma unroll
    for (int j = 0; j < 8; ++j) {
        int kk = kc * 32 + (hi << 4) + (j << 1);
        float f0 = (mv && (kk     < K)) ? wrow[kk]     : 0.0f;
        float f1 = (mv && (kk + 1 < K)) ? wrow[kk + 1] : 0.0f;
        union { __bf16 b[2]; unsigned int u; } pk;
        pk.b[0] = (__bf16)f0;
        pk.b[1] = (__bf16)f1;
        o[j] = pk.u;
    }
    uint4* dst = (uint4*)apack + (size_t)tid * 2;
    dst[0] = make_uint4(o[0], o[1], o[2], o[3]);
    dst[1] = make_uint4(o[4], o[5], o[6], o[7]);
}

// ---------------------------------------------------------------------------
// Implicit-GEMM dilated 3x3 conv, bf16 WMMA 16x16x32, f32 accumulate.
// Block = 8 waves = (8/NST) m-wave groups x NST 16-pixel strips.
// Each wave owns MPW M-tiles (MPW WMMAs sharing one B fragment per chunk).
// B-tiles (32k x 16px bf16) built cooperatively into LDS (double buffered);
// A-fragments pre-packed in global (2x b128 per tile per chunk, coalesced).
// 2-deep software pipeline: at iter kc the registers hold chunk kc+1;
// order = frag loads -> ds_store(kc+1) -> global loads(kc+2) -> WMMAs ->
// barrier, so the newest outstanding loads at the WMMA are chunk kc+2's and
// can remain in flight. No early-exit: EXEC is all-1s for every WMMA.
// ---------------------------------------------------------------------------
template <int NST, int MPW>
__global__ __launch_bounds__(256) void conv3_wmma(
    const unsigned int* __restrict__ apack,
    const float* __restrict__ x, int xCtot, int cin0, int Cin,
    const float* __restrict__ bias,
    float* __restrict__ y, int yCtot, int coff, int Cout, int dil, int nkc)
{
    constexpr int NMW = 8 / NST;     // m-wave groups per block
    constexpr int TPS = 256 / NST;   // producer threads per strip
    constexpr int GPT = 128 / TPS;   // groups per thread

    __shared__ __align__(16) unsigned int ldsB[2][NST][256]; // 2 bufs x strips x 1KB

    const int tid  = threadIdx.x;
    const int lane = tid & 31;
    const int wav  = tid >> 5;
    const int hi   = (lane >> 4) & 1;
    const int lo   = lane & 15;

    const int nXg = W_ / (16 * NST);
    int blk = blockIdx.x;
    const int xg = blk % nXg;  blk /= nXg;
    const int h  = blk % H_;   blk /= H_;
    const int b  = blk;
    const int w0b = xg * (16 * NST);

    const int K = Cin * 9;
    const float* xb = x + (size_t)(b * xCtot + cin0) * HW;

    // ---- producer per-thread constants (group g -> k-pair, n-pair) ----
    const int sp   = tid / TPS;
    const int tin  = tid % TPS;
    const int w0sp = w0b + sp * 16;

    int k0c[GPT], pixb[GPT], lds0[GPT], cia[GPT], ra[GPT];
#pragma unroll
    for (int t = 0; t < GPT; ++t) {
        int g  = tin + t * TPS;
        int kp = g >> 3;
        int n0 = (g & 7) << 1;
        k0c[t]  = kp << 1;                         // chunk-local even k
        pixb[t] = w0sp + n0;
        int lhalf = (k0c[t] >= 16) ? 16 : 0;
        lds0[t] = (n0 + lhalf) * 8 + ((k0c[t] & 15) >> 1);
        cia[t]  = k0c[t] / 9;                      // one-time divide
        ra[t]   = k0c[t] - cia[t] * 9;
    }

    // ---- consumer constants ----
    const int mw   = wav % NMW;
    const int sc   = wav / NMW;
    const int w0sc = w0b + sc * 16;

    const uint4* apcu[MPW];
#pragma unroll
    for (int u = 0; u < MPW; ++u)
        apcu[u] = (const uint4*)apack + ((size_t)(mw * MPW + u) * nkc * 32 + lane) * 2;

    const bool interior = (h - dil >= 0) && (h + dil < H_) &&
                          (w0b - dil >= 0) && (w0b + 16 * NST - 1 + dil < W_);
    const int nkc_full = ((K & 31) == 0) ? nkc : nkc - 1;  // chunks with no K-tail

    float va[GPT][4]; // (k,n0) (k,n0+1) (k+1,n0) (k+1,n0+1)

    auto pload_fast = [&]() {
#pragma unroll
        for (int t = 0; t < GPT; ++t) {
            int r0 = ra[t];
            int dy0 = r0 / 3 - 1, dx0 = r0 % 3 - 1;
            const float* p0 = xb + (size_t)cia[t] * HW + (h + dy0 * dil) * W_
                               + (pixb[t] + dx0 * dil);
            int r1 = r0 + 1, c1 = cia[t];
            if (r1 == 9) { r1 = 0; ++c1; }
            int dy1 = r1 / 3 - 1, dx1 = r1 % 3 - 1;
            const float* p1 = xb + (size_t)c1 * HW + (h + dy1 * dil) * W_
                               + (pixb[t] + dx1 * dil);
            va[t][0] = p0[0]; va[t][1] = p0[1];
            va[t][2] = p1[0]; va[t][3] = p1[1];
        }
    };

    auto pload_guard = [&](int kc) {
        const int kbase = kc * 32;
#pragma unroll
        for (int t = 0; t < GPT; ++t) {
            va[t][0] = va[t][1] = va[t][2] = va[t][3] = 0.0f;
            int kg = kbase + k0c[t];
            if (kg < K) {
                int r0 = ra[t];
                int dy0 = r0 / 3 - 1, dx0 = r0 % 3 - 1;
                int yy = h + dy0 * dil;
                int xx = pixb[t] + dx0 * dil;
                const float* p0 = xb + (size_t)cia[t] * HW + yy * W_ + xx;
                if ((unsigned)yy < (unsigned)H_) {
                    if ((unsigned)xx       < (unsigned)W_) va[t][0] = p0[0];
                    if ((unsigned)(xx + 1) < (unsigned)W_) va[t][1] = p0[1];
                }
                if (kg + 1 < K) {
                    int r1 = r0 + 1, c1 = cia[t];
                    if (r1 == 9) { r1 = 0; ++c1; }
                    int dy1 = r1 / 3 - 1, dx1 = r1 % 3 - 1;
                    int y1 = h + dy1 * dil;
                    int x1 = pixb[t] + dx1 * dil;
                    const float* p1 = xb + (size_t)c1 * HW + y1 * W_ + x1;
                    if ((unsigned)y1 < (unsigned)H_) {
                        if ((unsigned)x1       < (unsigned)W_) va[t][2] = p1[0];
                        if ((unsigned)(x1 + 1) < (unsigned)W_) va[t][3] = p1[1];
                    }
                }
            }
        }
    };

    auto padvance = [&]() {   // kg += 32  =>  ci += 3, r += 5 (32 = 3*9 + 5)
#pragma unroll
        for (int t = 0; t < GPT; ++t) {
            cia[t] += 3; ra[t] += 5;
            if (ra[t] >= 9) { ra[t] -= 9; ++cia[t]; }
        }
    };

    auto pstore = [&](int pb) {
        unsigned int* ls = &ldsB[pb][sp][0];
#pragma unroll
        for (int t = 0; t < GPT; ++t) {
            union { __bf16 bh[2]; unsigned int u; } q0, q1;
            q0.bh[0] = (__bf16)va[t][0]; q0.bh[1] = (__bf16)va[t][2];
            q1.bh[0] = (__bf16)va[t][1]; q1.bh[1] = (__bf16)va[t][3];
            ls[lds0[t]]     = q0.u;   // pixel n0
            ls[lds0[t] + 8] = q1.u;   // pixel n0+1 (merges to ds_store_2addr)
        }
    };

    f32x8 acc[MPW] = {};

    // ---- pipeline prologue: stage chunk 0 to LDS, preload chunk 1 ----
    if (interior && 0 < nkc_full) pload_fast(); else pload_guard(0);
    pstore(0);
    padvance();
    if (nkc > 1) {
        if (interior && 1 < nkc_full) pload_fast(); else pload_guard(1);
    }
    __syncthreads();

    int buf = 0;
    for (int kc = 0; kc < nkc; ++kc) {
        // 1) fragment loads (B from LDS[buf], A from packed global)
        union { uint4 u[2]; bf16x16 v; } au[MPW], bu;
        const uint4* bp = (const uint4*)&ldsB[buf][sc][0] + lane * 2;
        bu.u[0] = bp[0];
        bu.u[1] = bp[1];
#pragma unroll
        for (int u = 0; u < MPW; ++u) {
            const uint4* ap = apcu[u] + (size_t)kc * 64;
            au[u].u[0] = ap[0];
            au[u].u[1] = ap[1];
        }

        // 2) stage chunk kc+1 (already in registers) into the other buffer
        if (kc + 1 < nkc) pstore(buf ^ 1);

        // 3) issue chunk kc+2 input loads (stay in flight across the WMMAs)
        if (kc + 2 < nkc) {
            padvance();
            if (interior && (kc + 2 < nkc_full)) pload_fast();
            else                                 pload_guard(kc + 2);
        }

        // 4) matrix ops: MPW M-tiles share one B fragment
#pragma unroll
        for (int u = 0; u < MPW; ++u)
            acc[u] = __builtin_amdgcn_wmma_f32_16x16x32_bf16(
                false, au[u].v, false, bu.v, (short)0, acc[u], false, false);

        __syncthreads();
        buf ^= 1;
    }

    // ---- epilogue: C/D layout VGPR j = row j + 8*hi, lane lo = pixel col ----
#pragma unroll
    for (int u = 0; u < MPW; ++u) {
        float bl[8];
#pragma unroll
        for (int j = 0; j < 8; ++j) {
            int co = (mw * MPW + u) * 16 + j + hi * 8;
            bl[j] = (co < Cout) ? bias[co] : 0.0f;
        }
#pragma unroll
        for (int j = 0; j < 8; ++j) {
            int co = (mw * MPW + u) * 16 + j + hi * 8;
            if (co < Cout) {
                y[((size_t)(b * yCtot + coff + co)) * HW + h * W_ + w0sc + lo] =
                    acc[u][j] + bl[j];
            }
        }
    }
}

// ---------------------------------------------------------------------------
// BN batch statistics: per-(b, channel) plane partial sums -> f32 atomics.
// ---------------------------------------------------------------------------
__global__ __launch_bounds__(256) void bn_stats(
    const float* __restrict__ y, int Ctot, int c0, int C, float* __restrict__ stats)
{
    int cl = blockIdx.x % C;
    int b  = blockIdx.x / C;
    int c  = c0 + cl;
    const float4* p = (const float4*)(y + ((size_t)(b * Ctot + c)) * HW);
    float s = 0.0f, q = 0.0f;
    for (int i = threadIdx.x; i < HW / 4; i += 256) {
        float4 v = p[i];
        s += v.x + v.y + v.z + v.w;
        q += v.x * v.x + v.y * v.y + v.z * v.z + v.w * v.w;
    }
    __shared__ float sh_s[256];
    __shared__ float sh_q[256];
    sh_s[threadIdx.x] = s;
    sh_q[threadIdx.x] = q;
    __syncthreads();
    for (int o = 128; o > 0; o >>= 1) {
        if (threadIdx.x < (unsigned)o) {
            sh_s[threadIdx.x] += sh_s[threadIdx.x + o];
            sh_q[threadIdx.x] += sh_q[threadIdx.x + o];
        }
        __syncthreads();
    }
    if (threadIdx.x == 0) {
        atomicAdd(&stats[2 * c],     sh_s[0]);
        atomicAdd(&stats[2 * c + 1], sh_q[0]);
    }
}

// ---------------------------------------------------------------------------
// BN apply (training-mode biased var) + ReLU, optional residual add.
// Param channel = buffer channel + poff (handles interleaved concat layouts).
// ---------------------------------------------------------------------------
__global__ __launch_bounds__(256) void bn_apply(
    float* __restrict__ y, int Ctot, int c0, int C,
    const float* __restrict__ g, const float* __restrict__ be, int poff,
    const float* __restrict__ stats, const float* __restrict__ resid,
    float* __restrict__ dst)
{
    int cl = blockIdx.x % C;
    int b  = blockIdx.x / C;
    int c  = c0 + cl;
    const float invN = 1.0f / (float)((size_t)B_ * HW);
    float m  = stats[2 * c] * invN;
    float v  = stats[2 * c + 1] * invN - m * m;
    float rs = rsqrtf(v + EPS);
    float gg = g[c + poff] * rs;
    float bb = be[c + poff] - m * gg;
    size_t base = ((size_t)(b * Ctot + c)) * HW;
    float4*       py = (float4*)(y + base);
    float4*       pd = (float4*)(dst + base);
    const float4* pr = resid ? (const float4*)(resid + base) : nullptr;
    for (int i = threadIdx.x; i < HW / 4; i += 256) {
        float4 t = py[i];
        float4 o;
        o.x = fmaxf(t.x * gg + bb, 0.0f);
        o.y = fmaxf(t.y * gg + bb, 0.0f);
        o.z = fmaxf(t.z * gg + bb, 0.0f);
        o.w = fmaxf(t.w * gg + bb, 0.0f);
        if (pr) {
            float4 r = pr[i];
            o.x += r.x; o.y += r.y; o.z += r.z; o.w += r.w;
        }
        pd[i] = o;
    }
}

__global__ void zero_stats(float* stats, int n)
{
    int i = blockIdx.x * blockDim.x + threadIdx.x;
    if (i < n) stats[i] = 0.0f;
}

// ---------------------------------------------------------------------------
extern "C" void kernel_launch(void* const* d_in, const int* in_sizes, int n_in,
                              void* d_out, int out_size, void* d_ws, size_t ws_size,
                              hipStream_t stream)
{
    const float* x      = (const float*)d_in[0];
    const float* res_w  = (const float*)d_in[1];
    const float* res_b  = (const float*)d_in[2];
    const float* res_g  = (const float*)d_in[3];
    const float* res_be = (const float*)d_in[4];
    const float* in_w   = (const float*)d_in[5];
    const float* in_b   = (const float*)d_in[6];
    const float* in_g   = (const float*)d_in[7];
    const float* in_be  = (const float*)d_in[8];
    const float* s1_w   = (const float*)d_in[9];
    const float* s1_b   = (const float*)d_in[10];
    const float* bn1_g  = (const float*)d_in[11];
    const float* bn1_b  = (const float*)d_in[12];
    const float* s2_w   = (const float*)d_in[13];
    const float* s2_b   = (const float*)d_in[14];
    const float* bn2_g  = (const float*)d_in[15];
    const float* bn2_b  = (const float*)d_in[16];
    const float* s1d_w  = (const float*)d_in[17];
    const float* s1d_b  = (const float*)d_in[18];
    const float* bn1d_g = (const float*)d_in[19];
    const float* bn1d_b = (const float*)d_in[20];
    const float* out_w  = (const float*)d_in[21];
    const float* out_b  = (const float*)d_in[22];
    const float* out_g  = (const float*)d_in[23];
    const float* out_be = (const float*)d_in[24];
    float* out = (float*)d_out;

    const int DILS[4] = {1, 3, 5, 7};

    char* ws = (char*)d_ws;
    const size_t SZ64  = (size_t)B_ * 64  * HW * sizeof(float);  // 128 MiB
    const size_t SZ128 = (size_t)B_ * 128 * HW * sizeof(float);  // 256 MiB
    float* buf_res  = (float*)(ws);                              // [8, 64,HW] residual (post-BN)
    float* buf_main = (float*)(ws + SZ64);                       // [8,128,HW] hxin | scale1d
    float* buf_cat  = (float*)(ws + SZ64 + SZ128);               // [8,128,HW] [p1_i|p2_i] x4
    float* stats    = (float*)(ws + SZ64 + SZ128 + SZ128);       // 256 floats
    char*  apack_arena = ws + SZ64 + SZ128 + SZ128 + 4096;       // bf16 weight fragments
    size_t apoff = 0;

    auto conv = [&](const float* xin, int xCtot, int cin0, int Cin,
                    const float* wg, const float* bs, float* yo, int yCtot,
                    int coff, int Cout, int dil) {
        int K   = Cin * 9;
        int nkc = (K + 31) / 32;
        int tm  = (Cout + 15) / 16;       // 1 or 4
        unsigned int* ap = (unsigned int*)(apack_arena + apoff);
        apoff += (size_t)tm * nkc * 32 * 32;   // bytes: 32 lanes x 32B

        int total = tm * nkc * 32;
        hipLaunchKernelGGL(pack_w, dim3((total + 255) / 256), dim3(256), 0, stream,
                           wg, ap, Cout, K, nkc, total);

        if (tm == 4) {   // 2 m-wave groups (MPW=2) x 4 strips
            int blocks = B_ * H_ * (W_ / (16 * 4));
            hipLaunchKernelGGL((conv3_wmma<4, 2>), dim3(blocks), dim3(256), 0, stream,
                               ap, xin, xCtot, cin0, Cin, bs, yo, yCtot, coff, Cout, dil, nkc);
        } else {         // 1 m-wave group x 8 strips
            int blocks = B_ * H_ * (W_ / (16 * 8));
            hipLaunchKernelGGL((conv3_wmma<8, 1>), dim3(blocks), dim3(256), 0, stream,
                               ap, xin, xCtot, cin0, Cin, bs, yo, yCtot, coff, Cout, dil, nkc);
        }
    };
    auto statsz = [&]() {
        hipLaunchKernelGGL(zero_stats, dim3(1), dim3(256), 0, stream, stats, 256);
    };
    auto statsk = [&](const float* yo, int Ctot, int c0, int C) {
        hipLaunchKernelGGL(bn_stats, dim3(C * B_), dim3(256), 0, stream, yo, Ctot, c0, C, stats);
    };
    auto apply = [&](float* yo, int Ctot, int c0, int C,
                     const float* g, const float* be, int poff,
                     const float* resid, float* dst) {
        hipLaunchKernelGGL(bn_apply, dim3(C * B_), dim3(256), 0, stream,
                           yo, Ctot, c0, C, g, be, poff, stats, resid, dst);
    };

    // ---- residual = BNReLU(conv(x, res_w, d=1)) : 64 -> 64 ----
    conv(x, 64, 0, 64, res_w, res_b, buf_res, 64, 0, 64, 1);
    statsz(); statsk(buf_res, 64, 0, 64);
    apply(buf_res, 64, 0, 64, res_g, res_be, 0, nullptr, buf_res);

    // ---- hxin = BNReLU(conv(residual, in_w, d=1)) : 64 -> 64 (main ch 0-63) ----
    conv(buf_res, 64, 0, 64, in_w, in_b, buf_main, 128, 0, 64, 1);
    statsz(); statsk(buf_main, 128, 0, 64);
    apply(buf_main, 128, 0, 64, in_g, in_be, 0, nullptr, buf_main);

    // ---- scale1: 4 dilated branches 64 -> 16, part1_j at cat ch [32j, 32j+16) ----
    for (int j = 0; j < 4; ++j)
        conv(buf_main, 128, 0, 64, s1_w + (size_t)j * 16 * 64 * 9, s1_b + j * 16,
             buf_cat, 128, 32 * j, 16, DILS[j]);
    statsz();
    for (int j = 0; j < 4; ++j) statsk(buf_cat, 128, 32 * j, 16);
    for (int j = 0; j < 4; ++j)
        apply(buf_cat, 128, 32 * j, 16, bn1_g, bn1_b, -16 * j, nullptr, buf_cat);

    // ---- scale2: 16 branches 16 -> 4, block (i,j) at cat ch 32i+16+4j ----
    for (int i = 0; i < 4; ++i)
        for (int j = 0; j < 4; ++j)
            conv(buf_cat, 128, 32 * i, 16,
                 s2_w + (size_t)(4 * i + j) * 4 * 16 * 9, s2_b + (4 * i + j) * 4,
                 buf_cat, 128, 32 * i + 16 + 4 * j, 4, DILS[j]);
    statsz();
    for (int i = 0; i < 4; ++i) statsk(buf_cat, 128, 32 * i + 16, 16);
    for (int i = 0; i < 4; ++i)
        apply(buf_cat, 128, 32 * i + 16, 16, bn2_g, bn2_b, -(16 * i + 16), nullptr, buf_cat);

    // ---- scale1d: 4 branches concat(p1_i,p2_i) 32 -> 16, into main ch 64+16i ----
    for (int i = 0; i < 4; ++i)
        conv(buf_cat, 128, 32 * i, 32, s1d_w + (size_t)i * 16 * 32 * 9, s1d_b + i * 16,
             buf_main, 128, 64 + 16 * i, 16, DILS[i]);
    statsz(); statsk(buf_main, 128, 64, 64);
    apply(buf_main, 128, 64, 64, bn1d_g, bn1d_b, -64, nullptr, buf_main);

    // ---- hxout = BNReLU(conv(cat(hxin, scale1d), out_w, d=1)) + residual ----
    conv(buf_main, 128, 0, 128, out_w, out_b, out, 64, 0, 64, 1);
    statsz(); statsk(out, 64, 0, 64);
    apply(out, 64, 0, 64, out_g, out_be, 0, buf_res, out);
}
